// LSTMExtractor_9448928051783
// MI455X (gfx1250) — matrix operand order repaired and checked by reading, not compiled, not run on hardware
//
#include <hip/hip_runtime.h>
#include <cstdint>

#define B_   512
#define T_   512
#define D_   28
#define H_   128
#define BT_  16      // batch tile = WMMA M
#define THREADS_ 512 // 16 waves: waves 0-7 = layer 0, waves 8-15 = layer 1
#define WSTR 136     // padded LDS row stride (halves) for W_ih1: bank-conflict-free b128 reads
#define L2E_ 1.4426950408889634f

typedef __attribute__((ext_vector_type(16))) _Float16 v16h;
typedef __attribute__((ext_vector_type(8)))  float    v8f;
typedef __attribute__((ext_vector_type(4)))  uint32_t u32x4;
typedef __attribute__((ext_vector_type(2)))  uint32_t u32x2;

union HFrag  { v16h v; _Float16 h[16]; u32x4 q[2]; };
union HPack4 { _Float16 h[4]; u32x2 q; };

// Biased activations: bias folded into the exp2 argument via FMA (zero extra ops).
// sig_b(v, nb):  sigmoid(v + b)  with nb = -b * L2E
// tanh_b(v, pb): tanh(v + b)     with pb = 2 * b * L2E
__device__ __forceinline__ float sig_b(float v, float nb) {
  return __builtin_amdgcn_rcpf(1.0f + __builtin_amdgcn_exp2f(fmaf(v, -L2E_, nb)));
}
__device__ __forceinline__ float tanh_b(float v, float pb) {
  return fmaf(-2.0f,
              __builtin_amdgcn_rcpf(1.0f + __builtin_amdgcn_exp2f(fmaf(v, 2.0f * L2E_, pb))),
              1.0f);
}
__device__ __forceinline__ float fast_tanh(float v) {   // unbiased (cell tanh)
  return fmaf(-2.0f,
              __builtin_amdgcn_rcpf(1.0f + __builtin_amdgcn_exp2f(v * (2.0f * L2E_))),
              1.0f);
}

__device__ __forceinline__ v8f wmma_f16(v16h a, v16h b, v8f c) {
  return __builtin_amdgcn_wmma_f32_16x16x32_f16(false, a, false, b, (short)0, c,
                                                false, false);
}

// A fragment (16x32 f16, MxK) from row-major [16][H_] f16 LDS buffer.
__device__ __forceinline__ v16h load_A_lds(const _Float16* base, int lane, int kb) {
  const int m  = lane & 15;
  const int kh = lane >> 4;
  const _Float16* p = base + m * H_ + kb + kh * 8;
  HFrag f;
  f.q[0] = *(const u32x4*)(p);        // K offsets kh*8 + 0..7
  f.q[1] = *(const u32x4*)(p + 16);   // K offsets kh*8 + 16..23
  return f.v;
}

// B fragment (32x16 f16, KxN) from row-major f32 W[N][Kdim]; B[k][n] = W[n][k].
__device__ __forceinline__ v16h load_B_wt(const float* __restrict__ W, int Kdim,
                                          int lane, int nt, int kb) {
  const int n  = nt * 16 + (lane & 15);
  const int ks = kb + (lane >> 4) * 8;
  const float* row = W + (long)n * Kdim;
  HFrag f;
#pragma unroll
  for (int u = 0; u < 16; ++u) {
    const int k = ks + (u < 8 ? u : u + 8);
    f.h[u] = (_Float16)((k < Kdim) ? row[k] : 0.0f);
  }
  return f.v;
}

// B fragment from the padded f16 W_ih1 image in LDS ([512][WSTR]).
__device__ __forceinline__ v16h load_B_lds(const _Float16* Wl, int lane, int nt, int kb) {
  const int n  = nt * 16 + (lane & 15);
  const int ks = kb + (lane >> 4) * 8;
  const _Float16* p = Wl + n * WSTR + ks;   // n*272B (16B mult) + 16B mult
  HFrag f;
  f.q[0] = *(const u32x4*)(p);
  f.q[1] = *(const u32x4*)(p + 16);
  return f.v;
}

// B fragment of W_ih0 from the pre-swizzled f16 image in global workspace.
// Layout: frag nt (0..31), lane slice of 16 halves at (nt*32+lane)*16.
__device__ __forceinline__ v16h load_B_ws(const _Float16* __restrict__ Wf8,
                                          int lane, int nt) {
  const _Float16* p = Wf8 + ((long)nt * 32 + lane) * 16;
  HFrag f;
  f.q[0] = *(const u32x4*)(p);
  f.q[1] = *(const u32x4*)(p + 8);
  return f.v;
}

// A fragment (16x32, K zero-padded from D_=28) for layer-0 input x[:, t, :].
__device__ __forceinline__ v16h load_A_x(const float* __restrict__ x, int b0,
                                         int t, int lane) {
  const int m  = lane & 15;
  const int kh = lane >> 4;
  const float* row = x + ((long)(b0 + m) * T_ + t) * D_;
  HFrag f;
#pragma unroll
  for (int v = 0; v < 8; ++v) {
    const int k0 = (v < 4 ? 2 * v : 8 + 2 * v) + kh * 8;  // even; pairs never straddle 28
    float a0 = 0.0f, a1 = 0.0f;
    if (k0 < D_) { a0 = row[k0]; a1 = row[k0 + 1]; }
    f.h[2 * v]     = (_Float16)a0;
    f.h[2 * v + 1] = (_Float16)a1;
  }
  return f.v;
}

// ---- prep kernel: swizzle W_ih0 (f32 [512][28]) into f16 WMMA B-fragments in d_ws ----
__global__ __launch_bounds__(1024)
void prep_wih0_frags(const float* __restrict__ Wih0, _Float16* __restrict__ dst) {
  const int tid  = threadIdx.x;     // 1024 threads = 32 frags x 32 lanes
  const int lane = tid & 31;
  const int nt   = tid >> 5;
  const HFrag f = { load_B_wt(Wih0, D_, lane, nt, 0) };
  _Float16* p = dst + (long)tid * 16;
  *(u32x4*)(p)     = ((const HFrag*)&f)->q[0];
  *(u32x4*)(p + 8) = ((const HFrag*)&f)->q[1];
}

__global__ __launch_bounds__(THREADS_)
void lstm2_pipe_kernel(const float* __restrict__ x,
                       const float* __restrict__ Whh0,
                       const float* __restrict__ bih0, const float* __restrict__ bhh0,
                       const float* __restrict__ Wih1, const float* __restrict__ Whh1,
                       const float* __restrict__ bih1, const float* __restrict__ bhh1,
                       const _Float16* __restrict__ wih0f,   // swizzled frags in d_ws
                       float* __restrict__ out)
{
  __shared__ __align__(16) _Float16 h0buf[2][BT_ * H_];     // ping-pong layer-0 h, 8 KB
  __shared__ __align__(16) _Float16 h1buf[2][BT_ * H_];     // ping-pong layer-1 h, 8 KB
  __shared__ __align__(16) _Float16 wihl[4 * H_ * WSTR];    // f16 W_ih1 image, 136 KB

  const int  tid  = threadIdx.x;
  const int  lane = tid & 31;
  const int  wave = tid >> 5;
  const bool isB  = wave >= 8;                 // group B = layer 1
  const int  jt   = isB ? (wave - 8) : wave;   // gate-complete j-tile 0..7
  const int  b0   = blockIdx.x * BT_;

  // ---- cooperative, coalesced f32->f16 conversion of W_ih1 into LDS ----
  for (int e = tid; e < 4 * H_ * H_; e += THREADS_) {
    const int n = e >> 7, k = e & (H_ - 1);
    wihl[n * WSTR + k] = (_Float16)Wih1[e];
  }

  // ---- zero both ping-pong h states ----
  {
    HPack4 z; z.h[0] = z.h[1] = z.h[2] = z.h[3] = (_Float16)0.0f;
    *(u32x2*)&h0buf[0][tid * 4] = z.q;
    *(u32x2*)&h0buf[1][tid * 4] = z.q;
    *(u32x2*)&h1buf[0][tid * 4] = z.q;
    *(u32x2*)&h1buf[1][tid * 4] = z.q;
  }

  // ---- per-group recurrent weights in VGPRs (16 frags = 128 regs each) ----
  // group A: Wf = W_hh0; W_ih0 streams from d_ws (L2).
  // group B: Wf = W_hh1; W_ih1 streams from LDS.
  v16h  Wf[16];
  float sA0, sA1, sA3, tA2;   // activation constants with folded bias
  {
    const float* Wr = isB ? Whh1 : Whh0;
    const float* ba = isB ? bih1 : bih0;
    const float* bb = isB ? bhh1 : bhh0;
#pragma unroll
    for (int g = 0; g < 4; ++g)
#pragma unroll
      for (int kc = 0; kc < 4; ++kc)
        Wf[g * 4 + kc] = load_B_wt(Wr, H_, lane, g * 8 + jt, kc * 32);
    const int nb = jt * 16 + (lane & 15);
    const float b0v = ba[nb]           + bb[nb];            // gate i
    const float b1v = ba[nb + 128]     + bb[nb + 128];      // gate f
    const float b2v = ba[nb + 256]     + bb[nb + 256];      // gate g
    const float b3v = ba[nb + 384]     + bb[nb + 384];      // gate o
    sA0 = -b0v * L2E_;
    sA1 = -b1v * L2E_;
    tA2 =  b2v * (2.0f * L2E_);
    sA3 = -b3v * L2E_;
  }

  // c-state: lane owns cells (m = em+r, j = ej), r = 0..7 (group A: c0, group B: c1)
  const int em = 8 * (lane >> 4);
  const int ej = jt * 16 + (lane & 15);
  float cst[8];
#pragma unroll
  for (int r = 0; r < 8; ++r) cst[r] = 0.0f;
  __syncthreads();

  // ---- software-pipelined recurrence: 1 barrier per step ----
  // iteration s: group A does layer0 step s (s<T), group B does layer1 step s-1 (s>=1).
  // Everyone reads state [p], writes state [1-p], p = s&1: race-free by construction.
  for (int s = 0; s <= T_; ++s) {
    const int p = s & 1;
    if (!isB) {
      if (s < T_) {
        // stream x-projection operands first (global: longest latency)
        v16h Bx[4];
#pragma unroll
        for (int g = 0; g < 4; ++g) Bx[g] = load_B_ws(wih0f, lane, g * 8 + jt);
        const v16h Ax = load_A_x(x, b0, s, lane);

        // first WMMA of each chain takes C = 0 (inline constant; no acc init movs)
        v8f acc[4];
#pragma unroll
        for (int g = 0; g < 4; ++g) {
          const v8f z = {};
          acc[g] = wmma_f16(Ax, Bx[g], z);
        }
#pragma unroll
        for (int kc = 0; kc < 4; ++kc) {
          const v16h Ah = load_A_lds(&h0buf[p][0], lane, kc * 32);
#pragma unroll
          for (int g = 0; g < 4; ++g)
            acc[g] = wmma_f16(Ah, Wf[g * 4 + kc], acc[g]);
        }
        if (s + 1 < T_) {   // warm L2 with next step's x rows
          const float* nrow = x + ((long)(b0 + (lane & 15)) * T_ + (s + 1)) * D_;
          __builtin_prefetch(nrow, 0, 1);
        }
        // in-register LSTM update; bias applied inside the activations via FMA
        _Float16* dst = &h0buf[1 - p][0];
#pragma unroll
        for (int r = 0; r < 8; ++r) {
          const float iv = sig_b (acc[0][r], sA0);
          const float fv = sig_b (acc[1][r], sA1);
          const float gv = tanh_b(acc[2][r], tA2);
          const float ov = sig_b (acc[3][r], sA3);
          const float cv = fv * cst[r] + iv * gv;
          cst[r] = cv;
          dst[(em + r) * H_ + ej] = (_Float16)(ov * fast_tanh(cv));
        }
      }
    } else {
      if (s >= 1) {
        v8f acc[4];
        {   // kc = 0 of the W_ih1 chain seeds the accumulators with C = 0
          const v16h Aa = load_A_lds(&h0buf[p][0], lane, 0);
#pragma unroll
          for (int g = 0; g < 4; ++g) {
            const v16h Bw = load_B_lds(wihl, lane, g * 8 + jt, 0);
            const v8f z = {};
            acc[g] = wmma_f16(Aa, Bw, z);
          }
        }
#pragma unroll
        for (int kc = 1; kc < 4; ++kc) {   // h_seq0[s-1] @ W_ih1^T  (B from LDS)
          const v16h Aa = load_A_lds(&h0buf[p][0], lane, kc * 32);
#pragma unroll
          for (int g = 0; g < 4; ++g) {
            const v16h Bw = load_B_lds(wihl, lane, g * 8 + jt, kc * 32);
            acc[g] = wmma_f16(Aa, Bw, acc[g]);
          }
        }
#pragma unroll
        for (int kc = 0; kc < 4; ++kc) {   // h1[s-2] @ W_hh1^T  (B in VGPRs)
          const v16h Ab = load_A_lds(&h1buf[p][0], lane, kc * 32);
#pragma unroll
          for (int g = 0; g < 4; ++g)
            acc[g] = wmma_f16(Ab, Wf[g * 4 + kc], acc[g]);
        }
        _Float16* dst = &h1buf[1 - p][0];
        float hv[8];
#pragma unroll
        for (int r = 0; r < 8; ++r) {
          const float iv = sig_b (acc[0][r], sA0);
          const float fv = sig_b (acc[1][r], sA1);
          const float gv = tanh_b(acc[2][r], tA2);
          const float ov = sig_b (acc[3][r], sA3);
          const float cv = fv * cst[r] + iv * gv;
          cst[r] = cv;
          hv[r] = ov * fast_tanh(cv);
          dst[(em + r) * H_ + ej] = (_Float16)hv[r];
        }
        if (s == T_) {   // final hidden of the top layer -> output
#pragma unroll
          for (int r = 0; r < 8; ++r)
            out[(long)(b0 + em + r) * H_ + ej] = hv[r];
        }
      }
    }
    __syncthreads();
  }
}

extern "C" void kernel_launch(void* const* d_in, const int* in_sizes, int n_in,
                              void* d_out, int out_size, void* d_ws, size_t ws_size,
                              hipStream_t stream) {
  (void)in_sizes; (void)n_in; (void)out_size; (void)ws_size;
  const float* x    = (const float*)d_in[0];
  const float* Wih0 = (const float*)d_in[1];
  const float* Whh0 = (const float*)d_in[2];
  const float* bih0 = (const float*)d_in[3];
  const float* bhh0 = (const float*)d_in[4];
  const float* Wih1 = (const float*)d_in[5];
  const float* Whh1 = (const float*)d_in[6];
  const float* bih1 = (const float*)d_in[7];
  const float* bhh1 = (const float*)d_in[8];
  float*     out   = (float*)d_out;
  _Float16*  wih0f = (_Float16*)d_ws;    // 32 KB swizzled W_ih0 fragment image

  // 1) swizzle W_ih0 into WMMA fragment layout (32 KB, L2-resident)
  hipLaunchKernelGGL(prep_wih0_frags, dim3(1), dim3(1024), 0, stream, Wih0, wih0f);
  // 2) fused, layer-pipelined 2-layer LSTM
  hipLaunchKernelGGL(lstm2_pipe_kernel, dim3(B_ / BT_), dim3(THREADS_), 0, stream,
                     x, Whh0, bih0, bhh0, Wih1, Whh1, bih1, bhh1, wih0f, out);
}
